// MultiHeadAttention_23330262352475
// MI455X (gfx1250) — compile-verified, matrix operand order checked by reading, not run on hardware
//
#include <hip/hip_runtime.h>
#include <hip/hip_bf16.h>

// ---------------- CDNA5 WMMA / TDM types ----------------
typedef __attribute__((ext_vector_type(16))) __bf16 bf16x16;
typedef __attribute__((ext_vector_type(8)))  float  f32x8;
typedef __attribute__((ext_vector_type(4)))  unsigned int u32x4;
typedef __attribute__((ext_vector_type(8)))  int    i32x8;
typedef __attribute__((ext_vector_type(4)))  int    i32x4;

#define BATCH  2
#define SEQ    2048
#define DMODEL 1024
#define NHEAD  16
#define DHEAD  64

// ---------------- helpers ----------------
__device__ __forceinline__ unsigned short f2bf(float f) {
    unsigned int u = __float_as_uint(f);
    u += 0x7FFFu + ((u >> 16) & 1u);       // round-to-nearest-even
    return (unsigned short)(u >> 16);
}

// WMMA 16x16x32 bf16 fragment loader.
// Layout (ISA 7.12.2, 16-bit A 16x32): lane L holds row (L&15);
// elements 0..7  = K[ 8*(L>>4) ..  8*(L>>4)+7 ]
// elements 8..15 = K[16+8*(L>>4) .. 16+8*(L>>4)+7]
// BT (N x K row-major) uses the identical pattern with row = N index.
__device__ __forceinline__ bf16x16 load_frag(const unsigned short* base, int ld,
                                             int row_base, int k_base, int lane) {
    int r = lane & 15, h = lane >> 4;
    const unsigned short* p = base + (size_t)(row_base + r) * ld + k_base + h * 8;
    union { uint4 u[2]; bf16x16 v; } t;
    t.u[0] = *(const uint4*)(p);
    t.u[1] = *(const uint4*)(p + 16);
    return t.v;
}

__device__ __forceinline__ f32x8 wmma_bf16(bf16x16 a, bf16x16 b, f32x8 c) {
    return __builtin_amdgcn_wmma_f32_16x16x32_bf16(false, a, false, b,
                                                   (short)0, c, false, false);
}

// Generic->LDS byte offset: per ISA, LDS aperture physical addr = addr[31:0].
__device__ __forceinline__ unsigned lds_off(const void* p) {
    return (unsigned)(unsigned long long)p;
}

// ---- Tensor Data Mover: 2D bf16 tile load (rows x 32 cols) Global -> LDS ----
// D# per ISA 8.3/8.4: group0 = count/lds_addr/global_addr/type=2,
// group1 = data_size=2B, tensor dims, tile 32x256, row pitch. Groups 2/3 = 0 (2D).
__device__ __forceinline__ void tdm_load_bt(unsigned lds_addr,
                                            const unsigned short* gsrc,
                                            int pitch_elems) {
    unsigned long long ga = (unsigned long long)gsrc;
    u32x4 g0 = { 1u,                                   // count=1 (valid user D#)
                 lds_addr,                             // LDS byte address
                 (unsigned)(ga & 0xFFFFFFFFull),       // global_addr[31:0]
                 (unsigned)((ga >> 32) & 0x01FFFFFFull) | 0x80000000u }; // [56:32] | type=2
    i32x8 g1;
    g1[0] = 0x00010000;                                // data_size=1 (2 bytes), mask=0
    g1[1] = (pitch_elems & 0xFFFF) << 16;              // tensor_dim0[15:0]
    g1[2] = ((pitch_elems >> 16) & 0xFFFF) | (256 << 16); // dim0[31:16] | tensor_dim1=256
    g1[3] = (32 << 16);                                // tensor_dim1 hi | tile_dim0=32
    g1[4] = 256;                                       // tile_dim1=256, tile_dim2=0
    g1[5] = pitch_elems;                               // tensor_dim0_stride[31:0]
    g1[6] = 0;                                         // stride hi | dim1_stride lo
    g1[7] = 0;
    i32x4 z4 = { 0, 0, 0, 0 };
#if __clang_major__ >= 23
    i32x8 z8 = { 0, 0, 0, 0, 0, 0, 0, 0 };
    __builtin_amdgcn_tensor_load_to_lds(g0, g1, z4, z4, z8, 0);
#else
    __builtin_amdgcn_tensor_load_to_lds(g0, g1, z4, z4, 0);
#endif
}

// ---------------- f32 -> bf16 convert ----------------
__global__ void cvt_bf16_kernel(const float* __restrict__ in,
                                unsigned short* __restrict__ out, int n) {
    int i = blockIdx.x * blockDim.x + threadIdx.x;
    int stride = gridDim.x * blockDim.x;
    for (; i < n; i += stride) out[i] = f2bf(in[i]);
}

// ---------------- weight transpose + convert: WT[n][k] = bf16(W[k][n]) ----------------
__global__ void wtrans_kernel(const float* __restrict__ W,
                              unsigned short* __restrict__ WT) {
    __shared__ float tile[32][33];
    int kb = blockIdx.x * 32;   // K-dim base in W
    int nb = blockIdx.y * 32;   // N-dim base in W
    for (int i = threadIdx.y; i < 32; i += 8)
        tile[i][threadIdx.x] = W[(size_t)(kb + i) * DMODEL + nb + threadIdx.x];
    __syncthreads();
    for (int i = threadIdx.y; i < 32; i += 8)
        WT[(size_t)(nb + i) * DMODEL + kb + threadIdx.x] = f2bf(tile[threadIdx.x][i]);
}

// ---------------- generic WMMA GEMM: C = A(MxK) * B(KxN), BT given (NxK) ----------------
// B operand staged through LDS by the Tensor Data Mover, double-buffered:
// wave0 issues the next 256x32 bf16 tile (16KB) while all 4 waves compute on
// the current one. Sync = s_wait_tensorcnt(0) in wave0 + workgroup barriers.
// mode 0: bf16 out, head-split        Qh[b][h][s][d]
// mode 1: bf16 out, head-split+trans  VhT[b][h][d][s]
// mode 2: f32  out, row-major (MxN)
__global__ void gemm_bf16_kernel(const unsigned short* __restrict__ A,
                                 const unsigned short* __restrict__ BT,
                                 void* __restrict__ Cout,
                                 int M, int N, int K, int mode) {
    __shared__ unsigned short btile[2][256 * 32];   // 2 x 16KB
    int wave = threadIdx.x >> 5;
    int lane = threadIdx.x & 31;
    int m0   = blockIdx.x * 16;
    int nblk = blockIdx.y * 256;
    int n0w  = wave * 64;                            // this wave's slice in the LDS tile

    int nsteps = K / 32;
    if (wave == 0)
        tdm_load_bt(lds_off(&btile[0][0]), BT + (size_t)nblk * K, K);

    f32x8 acc[4] = {};
    for (int step = 0; step < nsteps; ++step) {
        int k = step * 32;
        if (wave == 0) __builtin_amdgcn_s_wait_tensorcnt(0);  // current tile landed
        __syncthreads();                                      // publish to all waves
        if (wave == 0 && step + 1 < nsteps)
            tdm_load_bt(lds_off(&btile[(step + 1) & 1][0]),
                        BT + (size_t)nblk * K + (k + 32), K);

        const unsigned short* lb = &btile[step & 1][0];
        bf16x16 a = load_frag(A, K, m0, k, lane);
        #pragma unroll
        for (int t = 0; t < 4; ++t) {
            bf16x16 b = load_frag(lb, 32, n0w + t * 16, 0, lane);
            acc[t] = wmma_bf16(a, b, acc[t]);
        }
        __syncthreads();                                      // WAR: done reading buffer
    }

    int lane15 = lane & 15, hi = lane >> 4;
    #pragma unroll
    for (int t = 0; t < 4; ++t) {
        int ncol = nblk + n0w + t * 16 + lane15;
        #pragma unroll
        for (int j = 0; j < 8; ++j) {
            int mrow = m0 + j + hi * 8;
            float val = acc[t][j];
            if (mode == 2) {
                ((float*)Cout)[(size_t)mrow * N + ncol] = val;
            } else {
                int b = mrow / SEQ, s = mrow % SEQ;
                int h = ncol / DHEAD, d = ncol % DHEAD;
                unsigned short bv = f2bf(val);
                if (mode == 0)
                    ((unsigned short*)Cout)[(((size_t)(b * NHEAD + h)) * SEQ + s) * DHEAD + d] = bv;
                else
                    ((unsigned short*)Cout)[(((size_t)(b * NHEAD + h)) * DHEAD + d) * SEQ + s] = bv;
            }
        }
    }
}

// ---------------- fused attention: logits -> softmax -> attn out -> attn @ V ----------------
// block = 128 threads (4 waves); one block handles (b,h, 16 query rows).
// LDS: 16x2048 f32 logits (128KB) + 16x2048 bf16 probs (64KB) = 192KB (WGP LDS = 320KB)
__global__ void attn_kernel(const unsigned short* __restrict__ Qh,
                            const unsigned short* __restrict__ Kh,
                            const unsigned short* __restrict__ VhT,
                            const float* __restrict__ mask,
                            float* __restrict__ attn_out,
                            unsigned short* __restrict__ ctx) {
    extern __shared__ char smem_raw[];
    float*          lf = (float*)smem_raw;                           // [16][2048]
    unsigned short* lp = (unsigned short*)(smem_raw + 16 * SEQ * 4); // [16][2048]

    int bh = blockIdx.y;
    int b  = bh / NHEAD;
    int q0 = blockIdx.x * 16;
    int wave = threadIdx.x >> 5;
    int lane = threadIdx.x & 31;
    int lane15 = lane & 15, hi = lane >> 4;

    const unsigned short* Qbase = Qh  + (size_t)bh * SEQ * DHEAD;
    const unsigned short* Kbase = Kh  + (size_t)bh * SEQ * DHEAD;
    const unsigned short* Vbase = VhT + (size_t)bh * DHEAD * SEQ;

    // ---- Phase 1: logits = Q K^T / sqrt(dh) + mask * -1e10 ----
    bf16x16 aq0 = load_frag(Qbase, DHEAD, q0, 0,  lane);
    bf16x16 aq1 = load_frag(Qbase, DHEAD, q0, 32, lane);

    for (int kt = 0; kt < 32; ++kt) {
        int kk = wave * 512 + kt * 16;                  // key-tile base
        bf16x16 bk0 = load_frag(Kbase, DHEAD, kk, 0,  lane);
        bf16x16 bk1 = load_frag(Kbase, DHEAD, kk, 32, lane);
        f32x8 acc = {};
        acc = wmma_bf16(aq0, bk0, acc);
        acc = wmma_bf16(aq1, bk1, acc);
        int col = kk + lane15;
        float mval = mask[(size_t)b * SEQ + col] * -1.0e10f;
        #pragma unroll
        for (int j = 0; j < 8; ++j) {
            int row = j + hi * 8;
            lf[row * SEQ + col] = acc[j] * 0.125f + mval;
        }
    }
    __syncthreads();

    // ---- Phase 2: row softmax (8 threads per row), write f32 attn + bf16 probs ----
    {
        int r   = threadIdx.x >> 3;    // 0..15
        int sub = threadIdx.x & 7;     // 8 lanes cooperate per row
        float mx = -3.0e38f;
        for (int i = 0; i < SEQ / 8; ++i) {
            int c = sub + 8 * i;
            mx = fmaxf(mx, lf[r * SEQ + c]);
        }
        for (int m = 1; m < 8; m <<= 1) mx = fmaxf(mx, __shfl_xor(mx, m, 8));

        float sum = 0.0f;
        for (int i = 0; i < SEQ / 8; ++i) {
            int c = sub + 8 * i;
            sum += __expf(lf[r * SEQ + c] - mx);
        }
        for (int m = 1; m < 8; m <<= 1) sum += __shfl_xor(sum, m, 8);
        float inv = 1.0f / sum;

        size_t arow = ((size_t)bh * SEQ + q0 + r) * SEQ;
        for (int i = 0; i < SEQ / 8; ++i) {
            int c = sub + 8 * i;
            float p = __expf(lf[r * SEQ + c] - mx) * inv;
            attn_out[arow + c] = p;
            lp[r * SEQ + c]    = f2bf(p);
        }
    }
    __syncthreads();

    // ---- Phase 3: ctx = attn @ V  (each wave does a 16-wide d-slice) ----
    int nb = wave * 16;
    f32x8 acc = {};
    for (int kk = 0; kk < SEQ; kk += 32) {
        bf16x16 ap = load_frag(lp,    SEQ, 0,  kk, lane);   // probs from LDS
        bf16x16 bv = load_frag(Vbase, SEQ, nb, kk, lane);   // VhT rows = d
        acc = wmma_bf16(ap, bv, acc);
    }
    int h = bh % NHEAD;
    int d = nb + lane15;
    #pragma unroll
    for (int j = 0; j < 8; ++j) {
        int s = q0 + j + hi * 8;
        ctx[((size_t)b * SEQ + s) * DMODEL + h * DHEAD + d] = f2bf(acc[j]);
    }
}

// ---------------- host launch ----------------
extern "C" void kernel_launch(void* const* d_in, const int* in_sizes, int n_in,
                              void* d_out, int out_size, void* d_ws, size_t ws_size,
                              hipStream_t stream) {
    const float* q    = (const float*)d_in[0];
    const float* k    = (const float*)d_in[1];
    const float* v    = (const float*)d_in[2];
    const float* mask = (const float*)d_in[3];
    const float* Wq   = (const float*)d_in[4];
    const float* Wk   = (const float*)d_in[5];
    const float* Wv   = (const float*)d_in[6];
    const float* Wo   = (const float*)d_in[7];

    float* out  = (float*)d_out;
    float* attn = out + (size_t)BATCH * SEQ * DMODEL;   // tuple: (out, attn)

    const size_t NE = (size_t)BATCH * SEQ * DMODEL;     // 4M activation elems
    const size_t NW = (size_t)DMODEL * DMODEL;          // 1M weight elems

    unsigned short* ws  = (unsigned short*)d_ws;
    unsigned short* qb  = ws;
    unsigned short* kb  = qb  + NE;
    unsigned short* vb  = kb  + NE;
    unsigned short* WqT = vb  + NE;
    unsigned short* WkT = WqT + NW;
    unsigned short* WvT = WkT + NW;
    unsigned short* WoT = WvT + NW;
    unsigned short* Qh  = WoT + NW;
    unsigned short* Kh  = Qh  + NE;
    unsigned short* VhT = Kh  + NE;
    unsigned short* ctx = VhT + NE;

    // 1) convert activations to bf16
    cvt_bf16_kernel<<<4096, 256, 0, stream>>>(q, qb, (int)NE);
    cvt_bf16_kernel<<<4096, 256, 0, stream>>>(k, kb, (int)NE);
    cvt_bf16_kernel<<<4096, 256, 0, stream>>>(v, vb, (int)NE);

    // 2) transpose+convert weights (BT layout for WMMA B operand)
    dim3 tg(DMODEL / 32, DMODEL / 32), tb(32, 8);
    wtrans_kernel<<<tg, tb, 0, stream>>>(Wq, WqT);
    wtrans_kernel<<<tg, tb, 0, stream>>>(Wk, WkT);
    wtrans_kernel<<<tg, tb, 0, stream>>>(Wv, WvT);
    wtrans_kernel<<<tg, tb, 0, stream>>>(Wo, WoT);

    // 3) projections (M=4096, N=1024, K=1024), TDM-staged weights
    dim3 gg(BATCH * SEQ / 16, DMODEL / 256);
    gemm_bf16_kernel<<<gg, 128, 0, stream>>>(qb, WqT, Qh,  BATCH * SEQ, DMODEL, DMODEL, 0);
    gemm_bf16_kernel<<<gg, 128, 0, stream>>>(kb, WkT, Kh,  BATCH * SEQ, DMODEL, DMODEL, 0);
    gemm_bf16_kernel<<<gg, 128, 0, stream>>>(vb, WvT, VhT, BATCH * SEQ, DMODEL, DMODEL, 1);

    // 4) fused attention (logits + softmax + attn writeback + attn@V)
    size_t smem = (size_t)16 * SEQ * 4 + (size_t)16 * SEQ * 2;   // 192 KB
    dim3 ag(SEQ / 16, BATCH * NHEAD);
    attn_kernel<<<ag, 128, smem, stream>>>(Qh, Kh, VhT, mask, attn, ctx);

    // 5) output projection -> f32 out
    gemm_bf16_kernel<<<gg, 128, 0, stream>>>(ctx, WoT, out, BATCH * SEQ, DMODEL, DMODEL, 2);
}
